// BruteForceMoE_24893630447780
// MI455X (gfx1250) — matrix operand order repaired and checked by reading, not compiled, not run on hardware
//
#include <hip/hip_runtime.h>
#include <math.h>

// MoE forward for MI455X (gfx1250, wave32, WMMA).
#define E      8
#define D      512
#define H      2048
#define TOPK   2
#define NTOK   1024
#define NROWS  (NTOK * TOPK)   // 2048 dispatched rows total (always)
#define CAP    NROWS           // worst-case rows a single expert can receive

typedef __attribute__((ext_vector_type(8)))  float  v8f;
typedef __attribute__((ext_vector_type(16))) __bf16 v16bf;

union Frag16 { uint4 q[2]; v16bf v; };

// fp32 -> bf16 (round-half-up; differs from RNE only on exact 0x8000 ties).
// One v_add per element, one v_perm_b32 per pair: 3 VALU per packed pair.
__device__ __forceinline__ unsigned bfround(float f) {
    return __float_as_uint(f) + 0x8000u;
}
__device__ __forceinline__ unsigned pk2(float x, float y) {
    // v_perm_b32: result bytes 0-3 select from src1, 4-7 from src0.
    // sel 0x07060302 -> { y[31:16], x[31:16] }
    return __builtin_amdgcn_perm(bfround(y), bfround(x), 0x07060302u);
}
__device__ __forceinline__ uint4 pack8(float4 a, float4 b) {
    uint4 r;
    r.x = pk2(a.x, a.y);
    r.y = pk2(a.z, a.w);
    r.z = pk2(b.x, b.y);
    r.w = pk2(b.z, b.w);
    return r;
}

__device__ __forceinline__ float gelu_erf(float x) {
    return 0.5f * x * (1.0f + erff(x * 0.70710678118654752f));
}

// ---------------------------------------------------------------- init
__global__ void init_counts(int* counts) {
    if (threadIdx.x < E) counts[threadIdx.x] = 0;
}

// ---------------------------------------------------------------- gate
__global__ __launch_bounds__(256) void gate_topk(
    const float* __restrict__ inp, const float* __restrict__ gate_w,
    const float* __restrict__ gate_b,
    int* __restrict__ counts, int* __restrict__ rowinfo,
    int* __restrict__ rowof, float* __restrict__ score)
{
    int n = blockIdx.x * blockDim.x + threadIdx.x;
    if (n >= NTOK) return;

    float acc[E];
#pragma unroll
    for (int e = 0; e < E; ++e) acc[e] = gate_b[e];

    const float4* x4 = (const float4*)(inp + (size_t)n * D);
    const float4* g4 = (const float4*)gate_w;
    for (int d = 0; d < D / 4; ++d) {
        float4 xv = x4[d];
#pragma unroll
        for (int e = 0; e < E; ++e) {
            float4 gv = g4[e * (D / 4) + d];
            acc[e] += xv.x * gv.x + xv.y * gv.y + xv.z * gv.z + xv.w * gv.w;
        }
    }

    // top-2, strict '>' so the lowest index wins ties (matches lax.top_k)
    float v0 = -3.402823466e+38f, v1 = -3.402823466e+38f;
    int   i0 = 0, i1 = 0;
#pragma unroll
    for (int e = 0; e < E; ++e) {
        float l = acc[e];
        if (l > v0)      { v1 = v0; i1 = i0; v0 = l; i0 = e; }
        else if (l > v1) { v1 = l;  i1 = e; }
    }
    float t  = expf(v1 - v0);            // <= 1
    float p0 = 1.0f / (1.0f + t);
    float p1 = t * p0;

    int s0 = atomicAdd(&counts[i0], 1);
    rowinfo[i0 * CAP + s0] = (n << 1);
    rowof[2 * n]           = (i0 << 11) | s0;
    score[2 * n]           = p0;

    int s1 = atomicAdd(&counts[i1], 1);
    rowinfo[i1 * CAP + s1] = (n << 1) | 1;
    rowof[2 * n + 1]       = (i1 << 11) | s1;
    score[2 * n + 1]       = p1;
}

// ---------------------------------------------------------------- scan
__global__ void scan_offsets(const int* __restrict__ counts, int* __restrict__ offsets) {
    if (threadIdx.x == 0) {
        int run = 0;
        for (int e = 0; e < E; ++e) { offsets[e] = run; run += counts[e]; }
    }
}

// -------------------------------------------------- WMMA fragment loads
// A (16x32 bf16, ISA 7.12.2): lane halves = K[koff..koff+7], K[16+koff..],
//   koff = 8*(lane>>4), row = lane&15.
// B (32x16 bf16): lane = N col, same K split as A:
//   halves = K[koff..koff+7], K[16+koff..], koff = 8*(lane>>4).
__device__ __forceinline__ void load_fragA(Frag16& f, const unsigned short (*T)[40],
                                           int row, int lane) {
    int koff = (lane >> 4) * 8;
    f.q[0] = *(const uint4*)&T[row][koff];
    f.q[1] = *(const uint4*)&T[row][16 + koff];
}
__device__ __forceinline__ void load_fragB(Frag16& f, const unsigned short (*T)[40],
                                           int col, int lane) {
    int koff = (lane >> 4) * 8;
    f.q[0] = *(const uint4*)&T[col][koff];
    f.q[1] = *(const uint4*)&T[col][16 + koff];
}

#define WMMA_BF16(A, B, C) \
    __builtin_amdgcn_wmma_f32_16x16x32_bf16(false, (A).v, false, (B).v, (short)0, (C), false, false)

// ---------------------------------------------------------------- GEMM1
// act[g][h] = GELU( x[tok(g)] . w1[e][h] + b1[e][h] ), stored bf16.
// Block tile 128x64, 8 waves, wave tile 32x32 (4 WMMA accs), double-buffered LDS.
__global__ __launch_bounds__(256) void moe_gemm1(
    const float* __restrict__ inp, const float* __restrict__ w1,
    const float* __restrict__ b1,
    const int* __restrict__ counts, const int* __restrict__ offsets,
    const int* __restrict__ rowinfo, unsigned short* __restrict__ act)
{
    const int e   = blockIdx.z;
    const int cnt = counts[e];
    const int m0  = blockIdx.y * 128;
    if (m0 >= cnt) return;                    // block-uniform early exit
    const int h0   = blockIdx.x * 64;
    const int offs = offsets[e];

    __shared__ unsigned short As[2][128][40]; // 2 x (128 rows x 32 K), pad 8
    __shared__ unsigned short Bs[2][64][40];

    const int tid = threadIdx.x;
    const int ldr = tid >> 2;                 // 0..63
    const int ldc = (tid & 3) * 8;            // 0,8,16,24

    int sA0 = m0 + ldr, sA1 = m0 + ldr + 64;
    int tok0 = (sA0 < cnt) ? (rowinfo[e * CAP + sA0] >> 1) : 0;
    int tok1 = (sA1 < cnt) ? (rowinfo[e * CAP + sA1] >> 1) : 0;
    const float* aSrc0 = inp + (size_t)tok0 * D;
    const float* aSrc1 = inp + (size_t)tok1 * D;
    const float* bSrc  = w1 + ((size_t)e * H + (size_t)(h0 + ldr)) * D;

    const int lane = tid & 31;
    const int wid  = tid >> 5;
    const int wm   = wid & 3;                 // 4 row-slabs of 32
    const int wn   = wid >> 2;                // 2 col-slabs of 32
    const int lr   = lane & 15;

    v8f acc00 = {}, acc01 = {}, acc10 = {}, acc11 = {};
    float4 ra0, ra1, ra2, ra3, rb0, rb1;

#define FETCH1(KB) {                                                     \
        const float4* p0 = (const float4*)(aSrc0 + (KB) + ldc);          \
        ra0 = p0[0]; ra1 = p0[1];                                        \
        const float4* p1 = (const float4*)(aSrc1 + (KB) + ldc);          \
        ra2 = p1[0]; ra3 = p1[1];                                        \
        const float4* pb = (const float4*)(bSrc + (KB) + ldc);           \
        rb0 = pb[0]; rb1 = pb[1]; }
#define STORE1(BUF) {                                                    \
        *(uint4*)&As[BUF][ldr][ldc]      = pack8(ra0, ra1);              \
        *(uint4*)&As[BUF][ldr + 64][ldc] = pack8(ra2, ra3);              \
        *(uint4*)&Bs[BUF][ldr][ldc]      = pack8(rb0, rb1); }

    FETCH1(0);
    STORE1(0);
    __syncthreads();

    const int nK = D / 32;
    for (int k = 0; k < nK; ++k) {
        const int  cur  = k & 1;
        const bool more = (k + 1) < nK;
        if (more) FETCH1((k + 1) * 32);

        Frag16 a0, a1, b0, b1f;
        load_fragA(a0,  As[cur], wm * 32 + lr,      lane);
        load_fragA(a1,  As[cur], wm * 32 + 16 + lr, lane);
        load_fragB(b0,  Bs[cur], wn * 32 + lr,      lane);
        load_fragB(b1f, Bs[cur], wn * 32 + 16 + lr, lane);
        acc00 = WMMA_BF16(a0, b0,  acc00);
        acc01 = WMMA_BF16(a0, b1f, acc01);
        acc10 = WMMA_BF16(a1, b0,  acc10);
        acc11 = WMMA_BF16(a1, b1f, acc11);

        if (more) STORE1(cur ^ 1);
        __syncthreads();
    }

    // C/D: lane -> N = lane&15 ; VGPR v -> M = v + 8*(lane>>4)
    const int rbase = m0 + wm * 32 + (lane >> 4) * 8;
    const int c0    = h0 + wn * 32 + lr;
    const float bias0 = b1[e * H + c0];
    const float bias1 = b1[e * H + c0 + 16];
#pragma unroll
    for (int v = 0; v < 8; ++v) {
        int s0r = rbase + v;
        if (s0r < cnt) {
            size_t g = (size_t)(offs + s0r) * H;
            act[g + c0]      = (unsigned short)(bfround(gelu_erf(acc00[v] + bias0)) >> 16);
            act[g + c0 + 16] = (unsigned short)(bfround(gelu_erf(acc01[v] + bias1)) >> 16);
        }
        int s1r = s0r + 16;
        if (s1r < cnt) {
            size_t g = (size_t)(offs + s1r) * H;
            act[g + c0]      = (unsigned short)(bfround(gelu_erf(acc10[v] + bias0)) >> 16);
            act[g + c0 + 16] = (unsigned short)(bfround(gelu_erf(acc11[v] + bias1)) >> 16);
        }
    }
#undef FETCH1
#undef STORE1
}

// ---------------------------------------------------------------- GEMM2
// y[g][d] = act[g] . w2[e][d] + b2[e][d]  (fp32 out)
__global__ __launch_bounds__(256) void moe_gemm2(
    const unsigned short* __restrict__ act, const float* __restrict__ w2,
    const float* __restrict__ b2,
    const int* __restrict__ counts, const int* __restrict__ offsets,
    float* __restrict__ y)
{
    const int e   = blockIdx.z;
    const int cnt = counts[e];
    const int m0  = blockIdx.y * 128;
    if (m0 >= cnt) return;
    const int d0   = blockIdx.x * 64;
    const int offs = offsets[e];

    __shared__ unsigned short As[2][128][40];
    __shared__ unsigned short Bs[2][64][40];

    const int tid = threadIdx.x;
    const int ldr = tid >> 2;
    const int ldc = (tid & 3) * 8;

    int sA0 = m0 + ldr, sA1 = m0 + ldr + 64;
    int g0r = offs + ((sA0 < cnt) ? sA0 : 0);
    int g1r = offs + ((sA1 < cnt) ? sA1 : 0);
    const unsigned short* aSrc0 = act + (size_t)g0r * H;
    const unsigned short* aSrc1 = act + (size_t)g1r * H;
    const float*          bSrc  = w2 + ((size_t)e * D + (size_t)(d0 + ldr)) * H;

    const int lane = tid & 31;
    const int wid  = tid >> 5;
    const int wm   = wid & 3;
    const int wn   = wid >> 2;
    const int lr   = lane & 15;

    v8f acc00 = {}, acc01 = {}, acc10 = {}, acc11 = {};
    uint4 qa0, qa1;
    float4 rb0, rb1;

#define FETCH2(KB) {                                                     \
        qa0 = *(const uint4*)(aSrc0 + (KB) + ldc);                       \
        qa1 = *(const uint4*)(aSrc1 + (KB) + ldc);                       \
        const float4* pb = (const float4*)(bSrc + (KB) + ldc);           \
        rb0 = pb[0]; rb1 = pb[1]; }
#define STORE2(BUF) {                                                    \
        *(uint4*)&As[BUF][ldr][ldc]      = qa0;                          \
        *(uint4*)&As[BUF][ldr + 64][ldc] = qa1;                          \
        *(uint4*)&Bs[BUF][ldr][ldc]      = pack8(rb0, rb1); }

    FETCH2(0);
    STORE2(0);
    __syncthreads();

    const int nK = H / 32;
    for (int k = 0; k < nK; ++k) {
        const int  cur  = k & 1;
        const bool more = (k + 1) < nK;
        if (more) FETCH2((k + 1) * 32);

        Frag16 a0, a1, b0, b1f;
        load_fragA(a0,  As[cur], wm * 32 + lr,      lane);
        load_fragA(a1,  As[cur], wm * 32 + 16 + lr, lane);
        load_fragB(b0,  Bs[cur], wn * 32 + lr,      lane);
        load_fragB(b1f, Bs[cur], wn * 32 + 16 + lr, lane);
        acc00 = WMMA_BF16(a0, b0,  acc00);
        acc01 = WMMA_BF16(a0, b1f, acc01);
        acc10 = WMMA_BF16(a1, b0,  acc10);
        acc11 = WMMA_BF16(a1, b1f, acc11);

        if (more) STORE2(cur ^ 1);
        __syncthreads();
    }

    const int rbase = m0 + wm * 32 + (lane >> 4) * 8;
    const int c0    = d0 + wn * 32 + lr;
    const float bias0 = b2[e * D + c0];
    const float bias1 = b2[e * D + c0 + 16];
#pragma unroll
    for (int v = 0; v < 8; ++v) {
        int s0r = rbase + v;
        if (s0r < cnt) {
            size_t g = (size_t)(offs + s0r) * D;
            y[g + c0]      = acc00[v] + bias0;
            y[g + c0 + 16] = acc01[v] + bias1;
        }
        int s1r = s0r + 16;
        if (s1r < cnt) {
            size_t g = (size_t)(offs + s1r) * D;
            y[g + c0]      = acc10[v] + bias0;
            y[g + c0 + 16] = acc11[v] + bias1;
        }
    }
#undef FETCH2
#undef STORE2
}

// ---------------------------------------------------------- combine + LN
__global__ __launch_bounds__(256) void combine_ln(
    const float* __restrict__ y, const int* __restrict__ rowof,
    const int* __restrict__ offsets, const float* __restrict__ score,
    const float* __restrict__ ln_w, const float* __restrict__ ln_b,
    float* __restrict__ out)
{
    int n  = blockIdx.x;
    int r0 = rowof[2 * n], r1 = rowof[2 * n + 1];
    int g0 = offsets[r0 >> 11] + (r0 & (CAP - 1));
    int g1 = offsets[r1 >> 11] + (r1 & (CAP - 1));
    float sc0 = score[2 * n], sc1 = score[2 * n + 1];

    int c0 = threadIdx.x, c1 = threadIdx.x + 256;
    float a = sc0 * y[(size_t)g0 * D + c0] + sc1 * y[(size_t)g1 * D + c0];
    float b = sc0 * y[(size_t)g0 * D + c1] + sc1 * y[(size_t)g1 * D + c1];

    float s  = a + b;
    float ss = a * a + b * b;
    for (int m = 16; m >= 1; m >>= 1) {
        s  += __shfl_xor(s,  m, 32);
        ss += __shfl_xor(ss, m, 32);
    }
    __shared__ float red[2][8];
    int lane = threadIdx.x & 31, wid = threadIdx.x >> 5;
    if (lane == 0) { red[0][wid] = s; red[1][wid] = ss; }
    __syncthreads();
    float ts = 0.f, tss = 0.f;
#pragma unroll
    for (int w = 0; w < 8; ++w) { ts += red[0][w]; tss += red[1][w]; }

    float mu  = ts * (1.0f / (float)D);
    float var = tss * (1.0f / (float)D) - mu * mu;
    float inv = rsqrtf(var + 1e-5f);
    out[(size_t)n * D + c0] = (a - mu) * inv * ln_w[c0] + ln_b[c0];
    out[(size_t)n * D + c1] = (b - mu) * inv * ln_w[c1] + ln_b[c1];
}

// ---------------------------------------------------------------- launch
extern "C" void kernel_launch(void* const* d_in, const int* in_sizes, int n_in,
                              void* d_out, int out_size, void* d_ws, size_t ws_size,
                              hipStream_t stream) {
    (void)in_sizes; (void)n_in; (void)out_size; (void)ws_size;
    const float* inp    = (const float*)d_in[0];
    const float* gate_w = (const float*)d_in[1];
    const float* gate_b = (const float*)d_in[2];
    const float* w1     = (const float*)d_in[3];
    const float* b1     = (const float*)d_in[4];
    const float* w2     = (const float*)d_in[5];
    const float* b2     = (const float*)d_in[6];
    const float* ln_w   = (const float*)d_in[7];
    const float* ln_b   = (const float*)d_in[8];
    float* out = (float*)d_out;

    // workspace layout (~12.7 MB total)
    char*  ws      = (char*)d_ws;
    int*   counts  = (int*)(ws + 0);                       // 8 ints
    int*   offsets = (int*)(ws + 64);                      // 8 ints
    float* score   = (float*)(ws + 128);                   // 2048 f32
    int*   rowof   = (int*)(ws + 128 + 8192);              // 2048 int
    int*   rowinfo = (int*)(ws + 128 + 8192 + 8192);       // E*CAP int = 64KB
    unsigned short* act = (unsigned short*)(ws + 82176);   // 2048x2048 bf16 = 8MB
    float* ybuf    = (float*)(ws + 82176 + (size_t)NROWS * H * 2);  // 2048x512 f32

    init_counts<<<1, 32, 0, stream>>>(counts);
    gate_topk<<<NTOK / 256, 256, 0, stream>>>(inp, gate_w, gate_b,
                                              counts, rowinfo, rowof, score);
    scan_offsets<<<1, 32, 0, stream>>>(counts, offsets);

    dim3 g1(H / 64, CAP / 128, E);
    moe_gemm1<<<g1, 256, 0, stream>>>(inp, w1, b1, counts, offsets, rowinfo, act);

    dim3 g2(D / 64, CAP / 128, E);
    moe_gemm2<<<g2, 256, 0, stream>>>(act, w2, b2, counts, offsets, ybuf);

    combine_ln<<<NTOK, 256, 0, stream>>>(ybuf, rowof, offsets, score, ln_w, ln_b, out);
}
